// DGACritic_12094627906172
// MI455X (gfx1250) — compile-verified, matrix operand order checked by reading, not run on hardware
//
#include <hip/hip_runtime.h>

// ---------------------------------------------------------------------------
// MI455X / gfx1250 fused DGA-critic forward.
// All GEMMs on v_wmma_f32_16x16x32_bf16 (wave32, 16x16 tiles, K=32).
// One wave owns 16 batch rows end-to-end; LDS used for C->A layout staging.
// Token inputs are staged by the Tensor Data Mover (tensor_load_to_lds,
// TENSORcnt) when the toolchain exposes the builtin.
// ---------------------------------------------------------------------------

typedef __bf16 bf16_t;
typedef __attribute__((ext_vector_type(16))) __bf16 v16bf;
typedef __attribute__((ext_vector_type(8)))  float  v8f;

#define N_AGENTS 8
#define S_DIM    48
#define A_DIM    16
#define D_DIM    128
#define H_DIM    256
#define B_TOTAL  131072
#define QK_SCALE 0.08838834764831845f   // 1/sqrt(128)

#if defined(__has_builtin)
#  if __has_builtin(__builtin_amdgcn_tensor_load_to_lds) && \
      __has_builtin(__builtin_amdgcn_s_wait_tensorcnt)
#    define USE_TDM 1
#  endif
#endif
#ifndef USE_TDM
#  define USE_TDM 0
#endif

// workspace layout (bf16 elements), all weights stored transposed: Wt[n][k]
#define WS_TOK   0                                   // 8 x [128][64]
#define WS_AQ    (WS_TOK  + N_AGENTS * D_DIM * 64)   // [128][128]
#define WS_AK    (WS_AQ   + D_DIM * D_DIM)
#define WS_AV    (WS_AK   + D_DIM * D_DIM)
#define WS_VQ    (WS_AV   + D_DIM * D_DIM)
#define WS_VK    (WS_VQ   + D_DIM * D_DIM)
#define WS_VV    (WS_VK   + D_DIM * D_DIM)
#define WS_GATE  (WS_VV   + D_DIM * D_DIM)           // [128][256]
#define WS_H1    (WS_GATE + D_DIM * 2 * D_DIM)       // [256][192]
#define WS_H2    (WS_H1   + H_DIM * (S_DIM + A_DIM + D_DIM)) // [256][256]

union AB16 { v16bf v; uint4 q[2]; bf16_t e[16]; };

#define WMMA_BF16(Af, Bf, Cv) \
  __builtin_amdgcn_wmma_f32_16x16x32_bf16(false, (Af), false, (Bf), (short)0, (Cv), false, false)

// ---- Tensor Data Mover: 2D fp32 tile (tw x th, row stride in elems) -> LDS -
#if USE_TDM
typedef unsigned int v4u __attribute__((ext_vector_type(4)));
typedef int          v4i __attribute__((ext_vector_type(4)));
typedef int          v8i_t __attribute__((ext_vector_type(8)));

__device__ __forceinline__ void tdm_load_2d(unsigned long long gaddr,
                                            unsigned int lds_byte,
                                            unsigned int tw, unsigned int th,
                                            unsigned int stride_elems) {
  // D# group 0: count=1 | lds_addr | global_addr[56:0] | type=2 (bits 127:126)
  v4u g0;
  g0.x = 1u;
  g0.y = lds_byte;
  g0.z = (unsigned int)gaddr;
  g0.w = ((unsigned int)(gaddr >> 32) & 0x01FFFFFFu) | 0x80000000u;
  // D# group 1: data_size=4B; tensor dims == tile dims; 48b dim0 stride
  v8i_t g1;
  g1[0] = (int)(2u << 16);                                   // data_size=2 (4B)
  g1[1] = (int)((tw & 0xFFFFu) << 16);                       // tensor_dim0[15:0]
  g1[2] = (int)(((tw >> 16) & 0xFFFFu) | ((th & 0xFFFFu) << 16)); // td0 hi | td1 lo
  g1[3] = (int)(((th >> 16) & 0xFFFFu) | ((tw & 0xFFFFu) << 16)); // td1 hi | tile_dim0
  g1[4] = (int)(th & 0xFFFFu);                               // tile_dim1 (dim2=0)
  g1[5] = (int)stride_elems;                                 // dim0_stride[31:0]
  g1[6] = 0;                                                 // stride hi | dim1_stride lo
  g1[7] = 0;
  v4i gz = {0, 0, 0, 0};
#if __clang_major__ >= 23
  v8i_t gz8 = {0, 0, 0, 0, 0, 0, 0, 0};
  __builtin_amdgcn_tensor_load_to_lds(g0, g1, gz, gz, gz8, 0);
#else
  __builtin_amdgcn_tensor_load_to_lds(g0, g1, gz, gz, 0);
#endif
}
#endif

// Stage agent's (s||a) 16-row slice into LDS: [16][48] fp32 then [16][16] fp32.
__device__ __forceinline__ void stage_token_input(const float* __restrict__ S,
                                                  const float* __restrict__ Ac,
                                                  float* sIn, int agent, int rowbase) {
#if USE_TDM
  // previous iteration's fragment reads must clear before DMA reuses buffer
  asm volatile("s_wait_dscnt 0" ::: "memory");
  tdm_load_2d((unsigned long long)(uintptr_t)(S + (size_t)rowbase * (N_AGENTS * S_DIM)
                                              + agent * S_DIM),
              (unsigned int)(uintptr_t)sIn, S_DIM, 16, N_AGENTS * S_DIM);
  tdm_load_2d((unsigned long long)(uintptr_t)(Ac + (size_t)rowbase * (N_AGENTS * A_DIM)
                                              + agent * A_DIM),
              (unsigned int)(uintptr_t)sIn + 16 * S_DIM * 4, A_DIM, 16, N_AGENTS * A_DIM);
  __builtin_amdgcn_s_wait_tensorcnt(0);
#else
  (void)S; (void)Ac; (void)sIn; (void)agent; (void)rowbase;
#endif
}

// ---- fragment builders -----------------------------------------------------
// A-layout (16-bit, 16x32): lane L holds row m = L&15; K groups
// {kb..kb+7} in VGPR0-3 and {kb+16..kb+23} in VGPR4-7, kb = 32*kt + 8*(L>>4).

__device__ __forceinline__ AB16 lds_afrag(const bf16_t* tile, int ldk, int kt, int lane) {
  const int m  = lane & 15;
  const int kb = kt * 32 + ((lane >> 4) << 3);
  AB16 f;
  f.q[0] = *(const uint4*)(tile + m * ldk + kb);
  f.q[1] = *(const uint4*)(tile + m * ldk + kb + 16);
  return f;
}

// B-layout mirrors A with M<->N on the transposed weight Wt[n][k].
__device__ __forceinline__ AB16 gbl_bfrag(const bf16_t* __restrict__ Wt, int K,
                                          int nt, int kt, int lane) {
  const int n  = nt * 16 + (lane & 15);
  const int kb = kt * 32 + ((lane >> 4) << 3);
  AB16 f;
  f.q[0] = *(const uint4*)(Wt + n * K + kb);
  f.q[1] = *(const uint4*)(Wt + n * K + kb + 16);
  return f;
}

__device__ __forceinline__ AB16 cvt8(const float4 p0, const float4 p1,
                                     AB16 f, int g) {
  f.e[g * 8 + 0] = (bf16_t)p0.x; f.e[g * 8 + 1] = (bf16_t)p0.y;
  f.e[g * 8 + 2] = (bf16_t)p0.z; f.e[g * 8 + 3] = (bf16_t)p0.w;
  f.e[g * 8 + 4] = (bf16_t)p1.x; f.e[g * 8 + 5] = (bf16_t)p1.y;
  f.e[g * 8 + 6] = (bf16_t)p1.z; f.e[g * 8 + 7] = (bf16_t)p1.w;
  return f;
}

// A-fragment of tok_in(agent) straight from global fp32 (every 8-wide K group
// stays on one side of the 48-boundary => two aligned float4 loads per group).
__device__ __forceinline__ AB16 tokin_afrag(const float* __restrict__ S,
                                            const float* __restrict__ Ac,
                                            int agent, int rowbase, int kt, int lane) {
  const int m  = rowbase + (lane & 15);
  const int kb = kt * 32 + ((lane >> 4) << 3);
  AB16 f;
#pragma unroll
  for (int g = 0; g < 2; ++g) {
    const int k0 = kb + g * 16;
    const float* src = (k0 < S_DIM)
        ? (S  + m * (N_AGENTS * S_DIM) + agent * S_DIM + k0)
        : (Ac + m * (N_AGENTS * A_DIM) + agent * A_DIM + (k0 - S_DIM));
    f = cvt8(*(const float4*)src, *(const float4*)(src + 4), f, g);
  }
  return f;
}

// Same, but from the TDM-staged LDS copy: [16][48] fp32 then [16][16] fp32.
__device__ __forceinline__ AB16 tokin_afrag_lds(const float* sIn, int kt, int lane) {
  const int m  = lane & 15;
  const int kb = kt * 32 + ((lane >> 4) << 3);
  AB16 f;
#pragma unroll
  for (int g = 0; g < 2; ++g) {
    const int k0 = kb + g * 16;
    const float* src = (k0 < S_DIM) ? (sIn + m * S_DIM + k0)
                                    : (sIn + 16 * S_DIM + m * A_DIM + (k0 - S_DIM));
    f = cvt8(*(const float4*)src, *(const float4*)(src + 4), f, g);
  }
  return f;
}

// Store one C tile (v8f, C layout: VGPR j -> m = j + 8*(lane>>4), n = lane&15)
// as bf16 into an LDS row-major tile.
__device__ __forceinline__ void store_c(bf16_t* tile, int ldk, int nt, int lane, v8f c) {
  const int n  = nt * 16 + (lane & 15);
  const int mh = (lane >> 4) << 3;
#pragma unroll
  for (int j = 0; j < 8; ++j) tile[(mh + j) * ldk + n] = (bf16_t)c[j];
}

// D = A(tile in LDS, 16xK) x Wt(N-major bf16, K x 16*NT), fp32 accumulate.
template <int KT, int NT>
__device__ __forceinline__ void gemm_lds(const bf16_t* tile, int ldk,
                                         const bf16_t* __restrict__ Wt, int K,
                                         int lane, v8f c[NT]) {
#pragma unroll
  for (int kt = 0; kt < KT; ++kt) {
    AB16 a = lds_afrag(tile, ldk, kt, lane);
#pragma unroll
    for (int nt = 0; nt < NT; ++nt) {
      AB16 b = gbl_bfrag(Wt, K, nt, kt, lane);
      c[nt] = WMMA_BF16(a.v, b.v, c[nt]);
    }
  }
}

__device__ __forceinline__ void add_bias(v8f c[8], const float* __restrict__ b,
                                         int nt0, int lane) {
#pragma unroll
  for (int nt = 0; nt < 8; ++nt) {
    const float bv = b[(nt0 + nt) * 16 + (lane & 15)];
#pragma unroll
    for (int j = 0; j < 8; ++j) c[nt][j] += bv;
  }
}

// token(agent) = relu(tok_in @ token_W[agent] + token_b[agent]) -> C regs
template <bool FROM_LDS>
__device__ __forceinline__ void compute_token(const float* __restrict__ S,
                                              const float* __restrict__ Ac,
                                              const float* sIn,
                                              const bf16_t* __restrict__ tokWt,
                                              const float* __restrict__ tokb,
                                              int agent, int rowbase, int lane,
                                              v8f c[8]) {
  AB16 a0, a1;
  if constexpr (FROM_LDS) {
    a0 = tokin_afrag_lds(sIn, 0, lane);
    a1 = tokin_afrag_lds(sIn, 1, lane);
  } else {
    a0 = tokin_afrag(S, Ac, agent, rowbase, 0, lane);
    a1 = tokin_afrag(S, Ac, agent, rowbase, 1, lane);
  }
  const bf16_t* Wt = tokWt + agent * (D_DIM * 64);
#pragma unroll
  for (int nt = 0; nt < 8; ++nt) {
    v8f acc = {0.f, 0.f, 0.f, 0.f, 0.f, 0.f, 0.f, 0.f};
    AB16 b0 = gbl_bfrag(Wt, 64, nt, 0, lane);
    AB16 b1 = gbl_bfrag(Wt, 64, nt, 1, lane);
    acc = WMMA_BF16(a0.v, b0.v, acc);
    acc = WMMA_BF16(a1.v, b1.v, acc);
    const float bv = tokb[agent * D_DIM + nt * 16 + (lane & 15)];
#pragma unroll
    for (int j = 0; j < 8; ++j) acc[j] = fmaxf(acc[j] + bv, 0.f);
    c[nt] = acc;
  }
}

// ---- weight prep: fp32 W[b][k][n] -> bf16 Wt[b][n][k] ----------------------
__global__ __launch_bounds__(256) void transpose_bf16(const float* __restrict__ W,
                                                      bf16_t* __restrict__ Wt,
                                                      int K, int N, int total) {
  int idx = blockIdx.x * blockDim.x + threadIdx.x;
  if (idx >= total) return;
  const int kn = K * N;
  const int b  = idx / kn;
  const int r  = idx - b * kn;
  const int n  = r / K;
  const int k  = r - n * K;
  Wt[(size_t)b * kn + (size_t)n * K + k] = (bf16_t)W[(size_t)b * kn + (size_t)k * N + n];
}

// ---- fused forward ---------------------------------------------------------
__global__ __launch_bounds__(128, 1) void critic_fused(
    const float* __restrict__ S, const float* __restrict__ Ac,
    const float* __restrict__ tokb,
    const float* __restrict__ aqb, const float* __restrict__ akb,
    const float* __restrict__ avb, const float* __restrict__ vqb,
    const float* __restrict__ vkb, const float* __restrict__ vvb,
    const float* __restrict__ gateb, const float* __restrict__ h1b,
    const float* __restrict__ h2b, const float* __restrict__ h3W,
    const float* __restrict__ h3b, const int* __restrict__ agent_idx,
    const bf16_t* __restrict__ ws, float* __restrict__ out) {
  __shared__ __align__(16) bf16_t sTok[4][16 * D_DIM];   // per-wave query / E tile
  __shared__ __align__(16) bf16_t sStg[4][16 * H_DIM];   // per-wave staging (K<=256)

  const int lane = threadIdx.x & 31;
  const int wave = threadIdx.x >> 5;
  const int rowbase = (blockIdx.x * 4 + wave) * 16;
  bf16_t* tokQ = sTok[wave];
  bf16_t* stg  = sStg[wave];
  // TDM input landing zone overlays the upper half of stg: during token
  // processing only stg[0 .. 16*128) is in use (ld = D_DIM tiles).
  float* sInW = (float*)(stg + 16 * D_DIM);   // 16*(48+16) fp32 = 4 KB

  __builtin_prefetch(S  + (size_t)rowbase * (N_AGENTS * S_DIM), 0, 1);
  __builtin_prefetch(Ac + (size_t)rowbase * (N_AGENTS * A_DIM), 0, 1);

  const int iq = *agent_idx;

  // ---- 1) query token -> LDS (bf16, 16x128) ----
  {
    v8f c[8];
    stage_token_input(S, Ac, sInW, iq, rowbase);
    compute_token<(bool)USE_TDM>(S, Ac, sInW, ws + WS_TOK, tokb, iq, rowbase, lane, c);
#pragma unroll
    for (int nt = 0; nt < 8; ++nt) store_c(tokQ, D_DIM, nt, lane, c[nt]);
  }

  // ---- 2) two attention groups: ally (tokens 1..3), adv (tokens 4..7) ----
  const bf16_t* Wq_tab[2] = {ws + WS_AQ, ws + WS_VQ};
  const bf16_t* Wk_tab[2] = {ws + WS_AK, ws + WS_VK};
  const bf16_t* Wv_tab[2] = {ws + WS_AV, ws + WS_VV};
  const float*  bq_tab[2] = {aqb, vqb};
  const float*  bk_tab[2] = {akb, vkb};
  const float*  bv_tab[2] = {avb, vvb};

  v8f hres[2][8];
#pragma unroll
  for (int g = 0; g < 2; ++g) {
    // q = query_token @ Wq + bq   (kept in C-layout regs)
    v8f q[8];
#pragma unroll
    for (int nt = 0; nt < 8; ++nt) q[nt] = (v8f){0.f,0.f,0.f,0.f,0.f,0.f,0.f,0.f};
    gemm_lds<4, 8>(tokQ, D_DIM, Wq_tab[g], D_DIM, lane, q);
    add_bias(q, bq_tab[g], 0, lane);

    // online softmax over tokens, accumulating mix = sum_t w_t * tok_t
    float rmax[8], rsum[8];
    v8f mix[8];
#pragma unroll
    for (int j = 0; j < 8; ++j) { rmax[j] = -1e30f; rsum[j] = 0.f; }
#pragma unroll
    for (int nt = 0; nt < 8; ++nt) mix[nt] = (v8f){0.f,0.f,0.f,0.f,0.f,0.f,0.f,0.f};

    const int t0 = g ? 4 : 1, t1 = g ? 8 : 4;
    for (int t = t0; t < t1; ++t) {
      v8f tk[8];
      stage_token_input(S, Ac, sInW, t, rowbase);
      compute_token<(bool)USE_TDM>(S, Ac, sInW, ws + WS_TOK, tokb, t, rowbase, lane, tk);
#pragma unroll
      for (int nt = 0; nt < 8; ++nt) store_c(stg, D_DIM, nt, lane, tk[nt]);

      v8f kk[8];
#pragma unroll
      for (int nt = 0; nt < 8; ++nt) kk[nt] = (v8f){0.f,0.f,0.f,0.f,0.f,0.f,0.f,0.f};
      gemm_lds<4, 8>(stg, D_DIM, Wk_tab[g], D_DIM, lane, kk);
      add_bias(kk, bk_tab[g], 0, lane);

      // logit[m] = SCALE * sum_n q[m,n]*k[m,n] ; reduce across 16-lane half
      float l[8];
#pragma unroll
      for (int j = 0; j < 8; ++j) l[j] = 0.f;
#pragma unroll
      for (int nt = 0; nt < 8; ++nt)
#pragma unroll
        for (int j = 0; j < 8; ++j) l[j] += q[nt][j] * kk[nt][j];
#pragma unroll
      for (int j = 0; j < 8; ++j) {
#pragma unroll
        for (int off = 1; off <= 8; off <<= 1) l[j] += __shfl_xor(l[j], off, 32);
        l[j] *= QK_SCALE;
      }
      // online-softmax update of (rmax, rsum, mix)
#pragma unroll
      for (int j = 0; j < 8; ++j) {
        const float nm   = fmaxf(rmax[j], l[j]);
        const float corr = __expf(rmax[j] - nm);
        const float wt   = __expf(l[j] - nm);
        rsum[j] = rsum[j] * corr + wt;
#pragma unroll
        for (int nt = 0; nt < 8; ++nt) mix[nt][j] = mix[nt][j] * corr + wt * tk[nt][j];
        rmax[j] = nm;
      }
    }
    // normalize and project once: h = (sum_t w_t tok_t) @ Wv + bv
#pragma unroll
    for (int j = 0; j < 8; ++j) {
      const float inv = 1.f / rsum[j];
#pragma unroll
      for (int nt = 0; nt < 8; ++nt) mix[nt][j] *= inv;
    }
#pragma unroll
    for (int nt = 0; nt < 8; ++nt) store_c(stg, D_DIM, nt, lane, mix[nt]);
#pragma unroll
    for (int nt = 0; nt < 8; ++nt) hres[g][nt] = (v8f){0.f,0.f,0.f,0.f,0.f,0.f,0.f,0.f};
    gemm_lds<4, 8>(stg, D_DIM, Wv_tab[g], D_DIM, lane, hres[g]);
    add_bias(hres[g], bv_tab[g], 0, lane);
  }

  // ---- 3) gate: z = sigmoid(concat(h_ally,h_adv) @ gate_W + b); E = lerp ----
  v8f E[8];
  {
#pragma unroll
    for (int nt = 0; nt < 8; ++nt) store_c(stg, 2 * D_DIM, nt, lane, hres[0][nt]);
#pragma unroll
    for (int nt = 0; nt < 8; ++nt) store_c(stg, 2 * D_DIM, nt + 8, lane, hres[1][nt]);
    v8f z[8];
#pragma unroll
    for (int nt = 0; nt < 8; ++nt) z[nt] = (v8f){0.f,0.f,0.f,0.f,0.f,0.f,0.f,0.f};
    gemm_lds<8, 8>(stg, 2 * D_DIM, ws + WS_GATE, 2 * D_DIM, lane, z);
    add_bias(z, gateb, 0, lane);
#pragma unroll
    for (int nt = 0; nt < 8; ++nt)
#pragma unroll
      for (int j = 0; j < 8; ++j) {
        const float zz = 1.f / (1.f + __expf(-z[nt][j]));
        E[nt][j] = zz * hres[0][nt][j] + (1.f - zz) * hres[1][nt][j];
      }
  }

  // ---- 4) MLP head ----
  // E -> tokQ tile (free now); first 64 K of sa_i come straight from global.
#pragma unroll
  for (int nt = 0; nt < 8; ++nt) store_c(tokQ, D_DIM, nt, lane, E[nt]);

  const bf16_t* WtH1 = ws + WS_H1;   // [256][192]
  const bf16_t* WtH2 = ws + WS_H2;   // [256][256]

  // h1 = relu(sa_i @ h1W + b), 16 n-tiles computed in two register halves
#pragma unroll
  for (int half = 0; half < 2; ++half) {
    v8f h1c[8];
#pragma unroll
    for (int nt = 0; nt < 8; ++nt) h1c[nt] = (v8f){0.f,0.f,0.f,0.f,0.f,0.f,0.f,0.f};
#pragma unroll
    for (int kt = 0; kt < 6; ++kt) {
      AB16 a = (kt < 2) ? tokin_afrag(S, Ac, iq, rowbase, kt, lane)
                        : lds_afrag(tokQ, D_DIM, kt - 2, lane);
#pragma unroll
      for (int nt = 0; nt < 8; ++nt) {
        AB16 b = gbl_bfrag(WtH1, S_DIM + A_DIM + D_DIM, half * 8 + nt, kt, lane);
        h1c[nt] = WMMA_BF16(a.v, b.v, h1c[nt]);
      }
    }
#pragma unroll
    for (int nt = 0; nt < 8; ++nt) {
      const float bv = h1b[(half * 8 + nt) * 16 + (lane & 15)];
#pragma unroll
      for (int j = 0; j < 8; ++j) h1c[nt][j] = fmaxf(h1c[nt][j] + bv, 0.f);
      store_c(stg, H_DIM, half * 8 + nt, lane, h1c[nt]);
    }
  }

  // h2 = relu(h1 @ h2W + b) fused with out = h2 . h3W
  float oacc[8];
#pragma unroll
  for (int j = 0; j < 8; ++j) oacc[j] = 0.f;
#pragma unroll
  for (int half = 0; half < 2; ++half) {
    v8f h2c[8];
#pragma unroll
    for (int nt = 0; nt < 8; ++nt) h2c[nt] = (v8f){0.f,0.f,0.f,0.f,0.f,0.f,0.f,0.f};
#pragma unroll
    for (int kt = 0; kt < 8; ++kt) {
      AB16 a = lds_afrag(stg, H_DIM, kt, lane);
#pragma unroll
      for (int nt = 0; nt < 8; ++nt) {
        AB16 b = gbl_bfrag(WtH2, H_DIM, half * 8 + nt, kt, lane);
        h2c[nt] = WMMA_BF16(a.v, b.v, h2c[nt]);
      }
    }
#pragma unroll
    for (int nt = 0; nt < 8; ++nt) {
      const float bv = h2b[(half * 8 + nt) * 16 + (lane & 15)];
      const float wv = h3W[(half * 8 + nt) * 16 + (lane & 15)];
#pragma unroll
      for (int j = 0; j < 8; ++j) oacc[j] += fmaxf(h2c[nt][j] + bv, 0.f) * wv;
    }
  }
  // reduce across the 16-lane half and write one float per row
#pragma unroll
  for (int j = 0; j < 8; ++j)
#pragma unroll
    for (int off = 1; off <= 8; off <<= 1) oacc[j] += __shfl_xor(oacc[j], off, 32);
  if ((lane & 15) == 0) {
    const float bo = h3b[0];
    const int mh = (lane >> 4) << 3;
#pragma unroll
    for (int j = 0; j < 8; ++j) out[rowbase + mh + j] = oacc[j] + bo;
  }
}

// ---------------------------------------------------------------------------
extern "C" void kernel_launch(void* const* d_in, const int* in_sizes, int n_in,
                              void* d_out, int out_size, void* d_ws, size_t ws_size,
                              hipStream_t stream) {
  (void)in_sizes; (void)n_in; (void)out_size; (void)ws_size;
  const float* S    = (const float*)d_in[0];
  const float* Ac   = (const float*)d_in[1];
  const float* tokW = (const float*)d_in[2];
  const float* tokb = (const float*)d_in[3];
  const float* aqW  = (const float*)d_in[4];
  const float* aqb  = (const float*)d_in[5];
  const float* akW  = (const float*)d_in[6];
  const float* akb  = (const float*)d_in[7];
  const float* avW  = (const float*)d_in[8];
  const float* avb  = (const float*)d_in[9];
  const float* vqW  = (const float*)d_in[10];
  const float* vqb  = (const float*)d_in[11];
  const float* vkW  = (const float*)d_in[12];
  const float* vkb  = (const float*)d_in[13];
  const float* vvW  = (const float*)d_in[14];
  const float* vvb  = (const float*)d_in[15];
  const float* gateW= (const float*)d_in[16];
  const float* gateb= (const float*)d_in[17];
  const float* h1W  = (const float*)d_in[18];
  const float* h1b  = (const float*)d_in[19];
  const float* h2W  = (const float*)d_in[20];
  const float* h2b  = (const float*)d_in[21];
  const float* h3W  = (const float*)d_in[22];
  const float* h3b  = (const float*)d_in[23];
  const int*   idx  = (const int*)d_in[24];
  float* out  = (float*)d_out;
  bf16_t* ws  = (bf16_t*)d_ws;

  auto xp = [&](const float* W, bf16_t* Wt, int K, int N, int batch) {
    const int total = batch * K * N;
    transpose_bf16<<<dim3((total + 255) / 256), dim3(256), 0, stream>>>(W, Wt, K, N, total);
  };
  xp(tokW,  ws + WS_TOK,  64,        D_DIM, N_AGENTS);
  xp(aqW,   ws + WS_AQ,   D_DIM,     D_DIM, 1);
  xp(akW,   ws + WS_AK,   D_DIM,     D_DIM, 1);
  xp(avW,   ws + WS_AV,   D_DIM,     D_DIM, 1);
  xp(vqW,   ws + WS_VQ,   D_DIM,     D_DIM, 1);
  xp(vkW,   ws + WS_VK,   D_DIM,     D_DIM, 1);
  xp(vvW,   ws + WS_VV,   D_DIM,     D_DIM, 1);
  xp(gateW, ws + WS_GATE, 2 * D_DIM, D_DIM, 1);
  xp(h1W,   ws + WS_H1,   S_DIM + A_DIM + D_DIM, H_DIM, 1);
  xp(h2W,   ws + WS_H2,   H_DIM,     H_DIM, 1);

  critic_fused<<<dim3(B_TOTAL / 64), dim3(128), 0, stream>>>(
      S, Ac, tokb, aqb, akb, avb, vqb, vkb, vvb, gateb, h1b, h2b,
      h3W, h3b, idx, ws, out);
}
